// CodeBook_34342558499367
// MI455X (gfx1250) — matrix operand order repaired and checked by reading, not compiled, not run on hardware
//
#include <hip/hip_runtime.h>

#define NN 32768
#define DD 512
#define KK 8192

typedef __attribute__((ext_vector_type(16))) __bf16 v16bf;
typedef __attribute__((ext_vector_type(8)))  float  v8f;

union BF16Frag { v16bf v; uint4 q[2]; };

__device__ inline unsigned short f2bf(float f) {
  unsigned u = __float_as_uint(f);
  u += 0x7FFFu + ((u >> 16) & 1u);          // round to nearest even
  return (unsigned short)(u >> 16);
}
__device__ inline float bf2f(unsigned short h) {
  return __uint_as_float(((unsigned)h) << 16);
}

// ---------------------------------------------------------------------------
// Kernel 1: L2-normalize rows (fp32) and split into bf16 hi/lo planes.
// One wave per row (16 elements per lane as 4x float4), 8 rows per block.
// ---------------------------------------------------------------------------
__global__ __launch_bounds__(256) void norm_split_kernel(
    const float* __restrict__ src, unsigned short* __restrict__ hi,
    unsigned short* __restrict__ lo)
{
  const int lane = threadIdx.x & 31;
  const int wave = threadIdx.x >> 5;
  const size_t row = (size_t)blockIdx.x * 8 + wave;

  const float4* p = (const float4*)(src + row * DD);
  float4 v[4];
  float s = 0.f;
#pragma unroll
  for (int i = 0; i < 4; ++i) {
    v[i] = p[lane + 32 * i];
    s += v[i].x * v[i].x + v[i].y * v[i].y + v[i].z * v[i].z + v[i].w * v[i].w;
  }
#pragma unroll
  for (int off = 16; off >= 1; off >>= 1) s += __shfl_xor(s, off, 32);
  const float scale = 1.f / fmaxf(sqrtf(s), 1e-12f);

  ushort4* ph = (ushort4*)(hi + row * DD);
  ushort4* pl = (ushort4*)(lo + row * DD);
#pragma unroll
  for (int i = 0; i < 4; ++i) {
    float c[4] = {v[i].x * scale, v[i].y * scale, v[i].z * scale, v[i].w * scale};
    ushort4 h4, l4;
    unsigned short* hh = &h4.x;
    unsigned short* ll = &l4.x;
#pragma unroll
    for (int j = 0; j < 4; ++j) {
      unsigned short hb = f2bf(c[j]);
      hh[j] = hb;
      ll[j] = f2bf(c[j] - bf2f(hb));
    }
    ph[lane + 32 * i] = h4;
    pl[lane + 32 * i] = l4;
  }
}

// ---------------------------------------------------------------------------
// Async staging helpers: GLOBAL_LOAD_ASYNC_TO_LDS_B128, tracked by ASYNCcnt.
// dst is a raw 32-bit LDS byte address (addr[31:0] of the flat pointer).
// ---------------------------------------------------------------------------
__device__ __forceinline__ void async_b128(unsigned lds_dst, const char* src) {
  asm volatile("global_load_async_to_lds_b128 %0, %1, off"
               :: "v"(lds_dst), "v"(src) : "memory");
}

// Stage one 64KB codebook chunk (32 codes x 512 K, hi+lo planes): 16 loads/thread.
__device__ __forceinline__ void issue_b_chunk(const char* cWh, const char* cWl,
                                              unsigned dstBase, int c0, int tid) {
#pragma unroll
  for (int j = 0; j < 16; ++j) {
    const int i = tid + 256 * j;                       // 0..4095
    const char* src = ((i >> 11) ? cWl : cWh)
                    + (size_t)c0 * 1024 + (size_t)(i & 2047) * 16;
    async_b128(dstBase + (unsigned)i * 16u, src);
  }
}

// ---------------------------------------------------------------------------
// Kernel 2: fused distance GEMM + argmax via bf16 hi/lo WMMA.
// Block = 256 threads = 8 waves, 4(M) x 2(N). 64 x-rows resident in LDS
// (hi+lo, 128 KB, staged async); codebook streamed in 32-code chunks through
// a double-buffered 2x64KB async LDS pipeline. 3 WMMAs (hh,hl,lh) per
// 32-wide K step give ~fp32 accuracy at bf16-WMMA rate.
// Dynamic LDS: 256 KB (CDNA5 WGP has 320 KB).
// ---------------------------------------------------------------------------
__global__ __launch_bounds__(256) void vq_argmax_kernel(
    const unsigned short* __restrict__ Xh, const unsigned short* __restrict__ Xl,
    const unsigned short* __restrict__ Wh, const unsigned short* __restrict__ Wl,
    int* __restrict__ idx_out)
{
  extern __shared__ unsigned short smem[];
  unsigned short* Ah = smem;                 // [64][512] hi
  unsigned short* Al = smem + 64 * DD;       // [64][512] lo
  unsigned short* B0 = smem + 2 * 64 * DD;   // [plane][32][512]
  unsigned short* B1 = B0 + 2 * 32 * DD;

  const unsigned ldsA  = (unsigned)(size_t)(void*)Ah;
  const unsigned ldsB0 = (unsigned)(size_t)(void*)B0;
  const unsigned ldsB1 = (unsigned)(size_t)(void*)B1;

  const int tid  = threadIdx.x;
  const int row0 = blockIdx.x * 64;

  const char* cWh = (const char*)Wh;
  const char* cWl = (const char*)Wl;

  // Async-stage the 64 A rows (hi+lo = 128 KB): 32 b128 loads per thread.
  {
    const char* cXh = (const char*)Xh + (size_t)row0 * 1024;
    const char* cXl = (const char*)Xl + (size_t)row0 * 1024;
#pragma unroll
    for (int j = 0; j < 32; ++j) {
      const int i = tid + 256 * j;                     // 0..8191
      const char* src = ((i >> 12) ? cXl : cXh) + (size_t)(i & 4095) * 16;
      async_b128(ldsA + (unsigned)i * 16u, src);
    }
  }
  // Prologue: stage first codebook chunk.
  issue_b_chunk(cWh, cWl, ldsB0, 0, tid);

  const int wave = tid >> 5;
  const int lane = tid & 31;
  const int mw   = wave >> 1;       // 0..3 : M tile
  const int nwv  = wave & 1;        // 0..1 : N tile
  const int lmod = lane & 15;
  const int g    = lane >> 4;       // half-wave group

  const unsigned short* arow_h = Ah + (size_t)(mw * 16 + lmod) * DD;
  const unsigned short* arow_l = Al + (size_t)(mw * 16 + lmod) * DD;

  float bestv[8];
  int   besti[8];
#pragma unroll
  for (int r = 0; r < 8; ++r) { bestv[r] = -3.0e38f; besti[r] = 0; }

  for (int ci = 0; ci < KK / 32; ++ci) {
    const int c0 = ci * 32;
    unsigned short* curB = (ci & 1) ? B1 : B0;

    if (ci + 1 < KK / 32) {
      // Overlap: issue next chunk into the other buffer, then retire all
      // older async work (A + current chunk); 16 newest stay in flight.
      issue_b_chunk(cWh, cWl, (ci & 1) ? ldsB0 : ldsB1, c0 + 32, tid);
      asm volatile("s_wait_asynccnt 0x10" ::: "memory");
    } else {
      asm volatile("s_wait_asynccnt 0x0" ::: "memory");
    }
    __syncthreads();                // current buffer visible block-wide

    const unsigned short* bph = curB + (size_t)(nwv * 16 + lmod) * DD;  // hi
    const unsigned short* bpl = bph + 32 * DD;                          // lo

    v8f acc = {0.f, 0.f, 0.f, 0.f, 0.f, 0.f, 0.f, 0.f};
#pragma unroll 4
    for (int kk = 0; kk < DD; kk += 32) {
      BF16Frag a_h, a_l, b_h, b_l;
      // A (16x32): halves 0..7 at K=kk+g*8, halves 8..15 at K=kk+16+g*8
      a_h.q[0] = *(const uint4*)(arow_h + kk + g * 8);
      a_h.q[1] = *(const uint4*)(arow_h + kk + 16 + g * 8);
      a_l.q[0] = *(const uint4*)(arow_l + kk + g * 8);
      a_l.q[1] = *(const uint4*)(arow_l + kk + 16 + g * 8);
      // B (32x16): halves 0..7 at K=kk+g*8, halves 8..15 at K=kk+16+g*8
      b_h.q[0] = *(const uint4*)(bph + kk + g * 8);
      b_h.q[1] = *(const uint4*)(bph + kk + 16 + g * 8);
      b_l.q[0] = *(const uint4*)(bpl + kk + g * 8);
      b_l.q[1] = *(const uint4*)(bpl + kk + 16 + g * 8);

      acc = __builtin_amdgcn_wmma_f32_16x16x32_bf16(false, a_h.v, false, b_h.v,
                                                    (short)0, acc, false, false);
      acc = __builtin_amdgcn_wmma_f32_16x16x32_bf16(false, a_h.v, false, b_l.v,
                                                    (short)0, acc, false, false);
      acc = __builtin_amdgcn_wmma_f32_16x16x32_bf16(false, a_l.v, false, b_h.v,
                                                    (short)0, acc, false, false);
    }

    // C layout: VGPR r, lane L -> M = r + 8*g (+ mw*16), N = L%16.
    // Codes ascend, so strict '>' keeps the lowest index on ties.
    const int code = c0 + nwv * 16 + lmod;
#pragma unroll
    for (int r = 0; r < 8; ++r) {
      float vv = acc[r];
      if (vv > bestv[r]) { bestv[r] = vv; besti[r] = code; }
    }
    __syncthreads();                // done reading before buffer reuse
  }

  // Cross-lane reduction over the 16 N-lanes sharing each row group.
#pragma unroll
  for (int r = 0; r < 8; ++r) {
#pragma unroll
    for (int off = 8; off >= 1; off >>= 1) {
      float ov = __shfl_xor(bestv[r], off, 32);
      int   oi = __shfl_xor(besti[r], off, 32);
      if (ov > bestv[r] || (ov == bestv[r] && oi < besti[r])) {
        bestv[r] = ov; besti[r] = oi;
      }
    }
  }

  // Cross-wave (nwv) reduction through LDS (reuse staging storage).
  float* red_v = (float*)smem;            // 64 rows x 2
  int*   red_i = (int*)smem + 128;
  if (lmod == 0) {
#pragma unroll
    for (int r = 0; r < 8; ++r) {
      int rl = mw * 16 + g * 8 + r;
      red_v[rl * 2 + nwv] = bestv[r];
      red_i[rl * 2 + nwv] = besti[r];
    }
  }
  __syncthreads();
  if (tid < 64) {
    float bv = red_v[tid * 2];
    int   bi = red_i[tid * 2];
    float ov = red_v[tid * 2 + 1];
    int   oi = red_i[tid * 2 + 1];
    if (ov > bv || (ov == bv && oi < bi)) { bv = ov; bi = oi; }
    idx_out[row0 + tid] = bi;
  }
}

// ---------------------------------------------------------------------------
// Kernel 3: gather + renormalize in fp32, straight-through output,
// per-row loss partial (deterministic), float-encoded index.
// ---------------------------------------------------------------------------
__global__ __launch_bounds__(256) void vq_output_kernel(
    const float* __restrict__ x, const float* __restrict__ cb,
    const int* __restrict__ idx, float* __restrict__ out,
    float* __restrict__ idx_out_f, float* __restrict__ partial)
{
  const int lane = threadIdx.x & 31;
  const int wave = threadIdx.x >> 5;
  const size_t row = (size_t)blockIdx.x * 8 + wave;
  const int code = idx[row];

  const float4* pw = (const float4*)(cb + (size_t)code * DD);
  const float4* px = (const float4*)(x + row * DD);
  float4 wv[4], xv[4];
  float sw = 0.f, sx = 0.f;
#pragma unroll
  for (int i = 0; i < 4; ++i) {
    wv[i] = pw[lane + 32 * i];
    xv[i] = px[lane + 32 * i];
    sw += wv[i].x * wv[i].x + wv[i].y * wv[i].y + wv[i].z * wv[i].z + wv[i].w * wv[i].w;
    sx += xv[i].x * xv[i].x + xv[i].y * xv[i].y + xv[i].z * xv[i].z + xv[i].w * xv[i].w;
  }
#pragma unroll
  for (int off = 16; off >= 1; off >>= 1) {
    sw += __shfl_xor(sw, off, 32);
    sx += __shfl_xor(sx, off, 32);
  }
  const float wsc = 1.f / fmaxf(sqrtf(sw), 1e-12f);
  const float xsc = 1.f / fmaxf(sqrtf(sx), 1e-12f);

  float d2 = 0.f;
  float4* po = (float4*)(out + row * DD);
#pragma unroll
  for (int i = 0; i < 4; ++i) {
    const float* wc = &wv[i].x;
    const float* xc = &xv[i].x;
    float4 o;
    float* oc = &o.x;
#pragma unroll
    for (int j = 0; j < 4; ++j) {
      float q  = wc[j] * wsc;
      float xn = xc[j] * xsc;
      float dq = q - xn;
      oc[j] = xn + dq;          // straight-through value, as in reference
      d2 += dq * dq;
    }
    po[lane + 32 * i] = o;
  }
#pragma unroll
  for (int off = 16; off >= 1; off >>= 1) d2 += __shfl_xor(d2, off, 32);
  if (lane == 0) {
    partial[row]   = d2;
    idx_out_f[row] = (float)code;
  }
}

// ---------------------------------------------------------------------------
// Kernel 4: deterministic fixed-order loss reduction.
// loss = 1.25 * mean(diff^2)
// ---------------------------------------------------------------------------
__global__ __launch_bounds__(256) void loss_reduce_kernel(
    const float* __restrict__ partial, float* __restrict__ loss_out)
{
  __shared__ float sm[256];
  float s = 0.f;
  for (int i = threadIdx.x; i < NN; i += 256) s += partial[i];
  sm[threadIdx.x] = s;
  __syncthreads();
#pragma unroll
  for (int off = 128; off >= 1; off >>= 1) {
    if (threadIdx.x < off) sm[threadIdx.x] += sm[threadIdx.x + off];
    __syncthreads();
  }
  if (threadIdx.x == 0)
    loss_out[0] = 1.25f * sm[0] / ((float)NN * (float)DD);
}

// ---------------------------------------------------------------------------
extern "C" void kernel_launch(void* const* d_in, const int* in_sizes, int n_in,
                              void* d_out, int out_size, void* d_ws, size_t ws_size,
                              hipStream_t stream) {
  const float* x  = (const float*)d_in[0];   // [N, D] fp32
  const float* cb = (const float*)d_in[1];   // [K, D] fp32
  float* out = (float*)d_out;                // [N*D] qst | [1] loss | [N] idx

  char* ws = (char*)d_ws;
  unsigned short* Xh = (unsigned short*)(ws);
  unsigned short* Xl = (unsigned short*)(ws + (size_t)NN * DD * 2);
  unsigned short* Wh = (unsigned short*)(ws + (size_t)NN * DD * 4);
  unsigned short* Wl = (unsigned short*)(ws + (size_t)NN * DD * 4 + (size_t)KK * DD * 2);
  int*   idx     = (int*)  (ws + (size_t)NN * DD * 4 + (size_t)KK * DD * 4);
  float* partial = (float*)(ws + (size_t)NN * DD * 4 + (size_t)KK * DD * 4 + (size_t)NN * 4);

  norm_split_kernel<<<NN / 8, 256, 0, stream>>>(x,  Xh, Xl);
  norm_split_kernel<<<KK / 8, 256, 0, stream>>>(cb, Wh, Wl);

  // 256 KB dynamic LDS: A hi/lo (128 KB) + double-buffered B chunks (2x64 KB).
  vq_argmax_kernel<<<NN / 64, 256, 262144, stream>>>(Xh, Xl, Wh, Wl, idx);

  float* loss_out = out + (size_t)NN * DD;
  float* idxf     = loss_out + 1;
  vq_output_kernel<<<NN / 8, 256, 0, stream>>>(x, cb, idx, out, idxf, partial);
  loss_reduce_kernel<<<1, 256, 0, stream>>>(partial, loss_out);
}
